// dendriticNet_11029476016394
// MI455X (gfx1250) — matrix operand order repaired and checked by reading, not compiled, not run on hardware
//
#include <hip/hip_runtime.h>

#define HDIM 256
#define BDIM 32768
#define MTILE 64

typedef __attribute__((ext_vector_type(16))) __bf16 v16bf;
typedef __attribute__((ext_vector_type(8)))  float v8f;

struct BFrag { uint4 lo, hi; };   // 32 bytes == v16bf

__device__ __forceinline__ unsigned short f2bf(float f) {
  unsigned u = __builtin_bit_cast(unsigned, f);
  u += 0x7FFFu + ((u >> 16) & 1u);          // round-to-nearest-even
  return (unsigned short)(u >> 16);
}
__device__ __forceinline__ float bf2f(unsigned short h) {
  unsigned u = ((unsigned)h) << 16;
  return __builtin_bit_cast(float, u);
}

// ---------------------------------------------------------------------------
// Pre-kernel: convert the 9 weight matrices to bf16 with folded scale factors.
// Output layout in d_ws: W[k] at k*65536 bf16 elements, k = usage order:
//   0:wpf0*DT*GB 1:wpi0*DT*GA 2:wpb0*DT*GA 3:wpf1*DT*GB 4:wpi1*DT*GA
//   5:wpb1*DT*GA 6:wpf2*DT*GB 7:wip0*DT*GD 8:wip1*DT*GD
// ---------------------------------------------------------------------------
struct WCvtArgs { const float* src[9]; float scale[9]; };

__global__ __launch_bounds__(256) void wcvt_kernel(WCvtArgs args,
                                                   unsigned short* __restrict__ dst) {
  int idx = blockIdx.x * blockDim.x + threadIdx.x;   // 0 .. 9*65536-1
  int w = idx >> 16;
  int e = idx & 65535;
  dst[idx] = f2bf(args.src[w][e] * args.scale[w]);
}

// ---------------------------------------------------------------------------
// Fused stepper: tanh -> LDS (bf16) -> 9 GEMMs via WMMA -> elementwise tail.
// One workgroup = 256 threads (8 waves) = 64 rows x 256 cols of output.
// ---------------------------------------------------------------------------
__global__ __launch_bounds__(256) void dnet_kernel(
    const float* __restrict__ data, const float* __restrict__ s0,
    const float* __restrict__ s1,   const float* __restrict__ s2,
    const float* __restrict__ i0,   const float* __restrict__ i1,
    const unsigned short* __restrict__ wbf, float* __restrict__ out) {

  // 6 activation tiles, bf16, row-major: tile t row r -> ldsA[t*64*128 + r*128 .. +127]
  __shared__ unsigned int ldsA[6 * MTILE * (HDIM / 2)];   // 192 KB
  __shared__ float nudge[2][MTILE];                       // DT*GSOM*mean(rs1/rs2)

  const int tid     = threadIdx.x;
  const int rowbase = blockIdx.x * MTILE;

  // ---- Stage: rho = tanh(x) -> bf16 into LDS (fused, no HBM intermediate) ----
  const float* srcs[6] = { data, s0, s1, s2, i0, i1 };
  for (int t = 0; t < 6; ++t) {
    const float4* src4 = (const float4*)(srcs[t] + (size_t)rowbase * HDIM);
    unsigned int* dst  = ldsA + t * MTILE * (HDIM / 2);
    #pragma unroll
    for (int it = 0; it < 16; ++it) {
      int f4 = it * 256 + tid;                 // 0..4095 float4s of the 64x256 tile
      float4 v = src4[f4];
      unsigned p0 = (unsigned)f2bf(tanhf(v.x)) | ((unsigned)f2bf(tanhf(v.y)) << 16);
      unsigned p1 = (unsigned)f2bf(tanhf(v.z)) | ((unsigned)f2bf(tanhf(v.w)) << 16);
      dst[f4 * 2 + 0] = p0;
      dst[f4 * 2 + 1] = p1;
    }
  }
  __syncthreads();

  // ---- Nudges: row means of rho(s1) (tile 2) and rho(s2) (tile 3) ----
  if (tid < 128) {
    int which = tid >> 6;                      // 0 -> rs1, 1 -> rs2
    int row   = tid & 63;
    const unsigned int* rp = ldsA + (2 + which) * MTILE * 128 + row * 128;
    float sum = 0.f;
    #pragma unroll 8
    for (int k = 0; k < 128; ++k) {
      unsigned u = rp[k];
      sum += bf2f((unsigned short)u) + bf2f((unsigned short)(u >> 16));
    }
    nudge[which][row] = (0.1f * 0.8f) * (sum * (1.0f / 256.0f));  // DT*GSOM*mean
  }
  __syncthreads();

  // ---- WMMA compute ----
  const int lane  = tid & 31;
  const int wave  = tid >> 5;
  const int msub  = wave & 3;          // which 16-row sub-tile (0..3)
  const int nhalf = wave >> 2;         // which half of the 16 n-tiles
  const int hi    = lane >> 4;         // lane group
  const int lx    = lane & 15;

  // output o = coef[o]*state[o] + sum_p ( A[pairA] @ W[pairW]^T ) (+ nudge)
  const int   npairs[5]   = { 3, 3, 1, 1, 1 };
  const int   pairA[5][3] = { {0,4,2}, {1,5,3}, {2,0,0}, {1,0,0}, {2,0,0} };
  const int   pairW[5][3] = { {0,1,2}, {3,4,5}, {6,0,0}, {7,0,0}, {8,0,0} };
  const float coef[5]     = { 0.81f, 0.81f, 0.89f, 0.81f, 0.81f };
  const float* stp[5]     = { s0, s1, s2, i0, i1 };

  for (int o = 0; o < 5; ++o) {
    for (int nt = 0; nt < 8; ++nt) {
      const int ntile = nhalf * 8 + nt;
      v8f acc = { 0.f, 0.f, 0.f, 0.f, 0.f, 0.f, 0.f, 0.f };

      for (int p = 0; p < npairs[o]; ++p) {
        // A fragment base: lane lx = M row, hi selects K 0..7 / 8..15 chunk
        const unsigned int* arow =
            ldsA + pairA[o][p] * MTILE * 128 + (msub * 16 + lx) * 128 + hi * 4;
        // B fragment base: lane lx = N col = W row; hi selects K 0..7 / 8..15 chunk
        const unsigned short* wrow =
            wbf + pairW[o][p] * 65536 + (ntile * 16 + lx) * 256 + hi * 8;

        #pragma unroll
        for (int kt = 0; kt < 8; ++kt) {       // K = 256 in steps of 32
          uint4 a0 = *(const uint4*)(arow + kt * 16);       // K chunk, bytes +0
          uint4 a1 = *(const uint4*)(arow + kt * 16 + 8);   // K chunk, bytes +32
          BFrag fa { a0, a1 };
          v16bf va = __builtin_bit_cast(v16bf, fa);

          uint4 b0 = *(const uint4*)(wrow + kt * 32);       // K chunk, bytes +0
          uint4 b1 = *(const uint4*)(wrow + kt * 32 + 16);
          BFrag fb { b0, b1 };
          v16bf vb = __builtin_bit_cast(v16bf, fb);

          acc = __builtin_amdgcn_wmma_f32_16x16x32_bf16(
              /*neg_a=*/false, va, /*neg_b=*/false, vb,
              /*c_mod=*/(short)0, acc, /*reuse_a=*/false, /*reuse_b=*/false);
        }
      }

      // ---- elementwise tail: out = coef*state + acc (+ nudge) ----
      const int col = ntile * 16 + lx;
      const int rl0 = msub * 16 + hi * 8;      // local row of acc[0] (C/D layout)
      const float* st = stp[o];
      const float  cf = coef[o];
      #pragma unroll
      for (int r = 0; r < 8; ++r) {
        const int rl   = rl0 + r;
        const int grow = rowbase + rl;
        float sv = st[(size_t)grow * HDIM + col];
        float ov = cf * sv + acc[r];
        if (o >= 3) ov += nudge[o - 3][rl];
        out[(size_t)o * BDIM * HDIM + (size_t)grow * HDIM + col] = ov;
      }
    }
  }
}

// ---------------------------------------------------------------------------
// in_sizes order: 0:data 1:s0 2:s1 3:s2 4:i0 5:i1
//                 6:wpf0 7:wpf1 8:wpf2 9:wpb0 10:wpb1 11:wip0 12:wip1 13:wpi0 14:wpi1
// ---------------------------------------------------------------------------
extern "C" void kernel_launch(void* const* d_in, const int* in_sizes, int n_in,
                              void* d_out, int out_size, void* d_ws, size_t ws_size,
                              hipStream_t stream) {
  (void)in_sizes; (void)n_in; (void)out_size; (void)ws_size;

  const float* data = (const float*)d_in[0];
  const float* s0   = (const float*)d_in[1];
  const float* s1   = (const float*)d_in[2];
  const float* s2   = (const float*)d_in[3];
  const float* i0   = (const float*)d_in[4];
  const float* i1   = (const float*)d_in[5];

  const float GBDT = 0.1f * 1.0f;   // DT*GB
  const float GADT = 0.1f * 0.8f;   // DT*GA
  const float GDDT = 0.1f * 1.0f;   // DT*GD

  WCvtArgs wa;
  wa.src[0] = (const float*)d_in[6];   wa.scale[0] = GBDT;  // wpf0
  wa.src[1] = (const float*)d_in[13];  wa.scale[1] = GADT;  // wpi0
  wa.src[2] = (const float*)d_in[9];   wa.scale[2] = GADT;  // wpb0
  wa.src[3] = (const float*)d_in[7];   wa.scale[3] = GBDT;  // wpf1
  wa.src[4] = (const float*)d_in[14];  wa.scale[4] = GADT;  // wpi1
  wa.src[5] = (const float*)d_in[10];  wa.scale[5] = GADT;  // wpb1
  wa.src[6] = (const float*)d_in[8];   wa.scale[6] = GBDT;  // wpf2
  wa.src[7] = (const float*)d_in[11];  wa.scale[7] = GDDT;  // wip0
  wa.src[8] = (const float*)d_in[12];  wa.scale[8] = GDDT;  // wip1

  unsigned short* wbf = (unsigned short*)d_ws;   // 9*65536 bf16 = 1.18 MB

  hipLaunchKernelGGL(wcvt_kernel, dim3((9 * 65536) / 256), dim3(256), 0, stream,
                     wa, wbf);
  hipLaunchKernelGGL(dnet_kernel, dim3(BDIM / MTILE), dim3(256), 0, stream,
                     data, s0, s1, s2, i0, i1, (const unsigned short*)wbf,
                     (float*)d_out);
}
